// RNNSpeechDecoderModel_15951508537990
// MI455X (gfx1250) — matrix operand order repaired and checked by reading, not compiled, not run on hardware
//
#include <hip/hip_runtime.h>

// ---------------------------------------------------------------------------
// Types for CDNA5 WMMA bf16 path; explicit global (AS1) pointers so the
// backend emits global_load/global_store (not flat_*) in the hot loops.
// ---------------------------------------------------------------------------
typedef __attribute__((ext_vector_type(16))) __bf16 bf16x16;
typedef __attribute__((ext_vector_type(8)))  __bf16 bf16x8;
typedef __attribute__((ext_vector_type(8)))  float  f32x8;

#define GLB __attribute__((address_space(1)))
typedef const __bf16 GLB* gcbf16p;
typedef       __bf16 GLB* gbf16p;
typedef const bf16x8 GLB* gcv8p;
typedef const float  GLB* gcfp;
typedef       float  GLB* gfp;
typedef const int    GLB* gcip;

// Build a 16-element bf16 A/B fragment from two 16-byte global loads.
__device__ __forceinline__ bf16x16 ldfrag(gcbf16p lo, gcbf16p hi) {
    bf16x8 l = *(gcv8p)lo;
    bf16x8 h = *(gcv8p)hi;
    bf16x16 r;
#pragma unroll
    for (int i = 0; i < 8; ++i) { r[i] = l[i]; r[i + 8] = h[i]; }
    return r;
}

// ---------------------------------------------------------------------------
// bf16 WMMA GEMM:  C[M,N] = A[M,K] * Bt[N,K]^T  (+ bias, + epilogue)
//   A_MODE 0: A row-major, leading dim lda
//   A_MODE 1: layer-0 window mode: A row m, col k -> hpad[m*4 + (k>>9)][k&511]
//             (K fixed at 32 frames * 512)
//   EPI 0: softsign(acc+bias) -> bf16 C     (day transform)
//   EPI 1: acc + bias         -> fp32 C     (GRU input gates)
// Tiles: block 128 threads = 4 waves (2x2), wave tile 64x64, K step 32.
// All offsets kept in unsigned 32-bit (matrices < 4GB) so loads lower to
// global_load_b128 saddr+voffset with 32-bit offset increments.
// ---------------------------------------------------------------------------
template<int A_MODE, int EPI>
__global__ __launch_bounds__(128) void k_gemm(
    const __bf16* __restrict__ A, const __bf16* __restrict__ Bt,
    const float* __restrict__ bias, const int* __restrict__ day_idx,
    void* __restrict__ Cout, int N, int K,
    int lda, int ldk)
{
    const gcbf16p Ag = (gcbf16p)A;
    const gcbf16p Bg = (gcbf16p)Bt;

    const int tid  = threadIdx.x;
    const int wave = tid >> 5;
    const int lane = tid & 31;
    const int half = lane >> 4;
    const int l16  = lane & 15;
    const int mb = blockIdx.x * 128 + (wave >> 1) * 64;
    const int nb = blockIdx.y * 128 + (wave & 1) * 64;

    f32x8 acc[4][4];
#pragma unroll
    for (int i = 0; i < 4; ++i)
#pragma unroll
        for (int j = 0; j < 4; ++j)
#pragma unroll
            for (int v = 0; v < 8; ++v) acc[i][j][v] = 0.0f;

    // B fragment offsets: lane covers K 8h..8h+7 and 16+8h..16+8h+7 at n-row of Bt.
    unsigned boff[4];
#pragma unroll
    for (int j = 0; j < 4; ++j)
        boff[j] = (unsigned)(nb + j * 16 + l16) * (unsigned)ldk + half * 8;

    if (A_MODE == 0) {
        unsigned aoff[4];
#pragma unroll
        for (int i = 0; i < 4; ++i)
            aoff[i] = (unsigned)(mb + i * 16 + l16) * (unsigned)lda + half * 8;

        for (int k0 = 0; k0 < K; k0 += 32) {
            bf16x16 af[4], bfr[4];
#pragma unroll
            for (int i = 0; i < 4; ++i) {
                af[i] = ldfrag(Ag + aoff[i], Ag + aoff[i] + 16);
                aoff[i] += 32;
            }
#pragma unroll
            for (int j = 0; j < 4; ++j) {
                bfr[j] = ldfrag(Bg + boff[j], Bg + boff[j] + 16);
                boff[j] += 32;
            }
#pragma unroll
            for (int i = 0; i < 4; ++i)
#pragma unroll
                for (int j = 0; j < 4; ++j)
                    acc[i][j] = __builtin_amdgcn_wmma_f32_16x16x32_bf16(
                        false, af[i], false, bfr[j], (short)0, acc[i][j], false, false);
        }
    } else {
        // Layer-0 im2col-free mode: K = 32 frames * 512 dims; A row m of the
        // virtual [2048 x 16384] matrix maps frame fr to hpad row m*4+fr.
        for (int fr = 0; fr < 32; ++fr) {
            unsigned aoff[4];
#pragma unroll
            for (int i = 0; i < 4; ++i)
                aoff[i] = (unsigned)((mb + i * 16 + l16) * 4 + fr) * 512u + half * 8;

            for (int kk = 0; kk < 512; kk += 32) {
                bf16x16 af[4], bfr[4];
#pragma unroll
                for (int i = 0; i < 4; ++i) {
                    af[i] = ldfrag(Ag + aoff[i], Ag + aoff[i] + 16);
                    aoff[i] += 32;
                }
#pragma unroll
                for (int j = 0; j < 4; ++j) {
                    bfr[j] = ldfrag(Bg + boff[j], Bg + boff[j] + 16);
                    boff[j] += 32;
                }
#pragma unroll
                for (int i = 0; i < 4; ++i)
#pragma unroll
                    for (int j = 0; j < 4; ++j)
                        acc[i][j] = __builtin_amdgcn_wmma_f32_16x16x32_bf16(
                            false, af[i], false, bfr[j], (short)0, acc[i][j], false, false);
            }
        }
    }

    unsigned bias_off = 0;
    if (day_idx) bias_off = (unsigned)(*(gcip)day_idx) * (unsigned)N;
    const gcfp biasg = (gcfp)bias + bias_off;

#pragma unroll
    for (int i = 0; i < 4; ++i)
#pragma unroll
        for (int j = 0; j < 4; ++j) {
            const int col = nb + j * 16 + l16;
            const float bv = biasg[col];
#pragma unroll
            for (int v = 0; v < 8; ++v) {
                const int row = mb + i * 16 + half * 8 + v;
                float val = acc[i][j][v] + bv;
                if (EPI == 0) {
                    val = val / (1.0f + fabsf(val));   // soft_sign
                    ((gbf16p)Cout)[(unsigned)row * (unsigned)N + col] = (__bf16)val;
                } else {
                    ((gfp)Cout)[(unsigned)row * (unsigned)N + col] = val;
                }
            }
        }
}

// ---------------------------------------------------------------------------
// fp32 -> bf16 convert (grid-stride)
// ---------------------------------------------------------------------------
__global__ __launch_bounds__(256) void k_cvt_bf16(const float* __restrict__ src,
                                                  __bf16* __restrict__ dst, long n)
{
    for (long i = blockIdx.x * 256L + threadIdx.x; i < n; i += gridDim.x * 256L)
        dst[i] = (__bf16)src[i];
}

// day_weights[day_idx] (512x512) -> bf16
__global__ __launch_bounds__(256) void k_cvt_day_w(const float* __restrict__ dayW,
                                                   const int* __restrict__ idx,
                                                   __bf16* __restrict__ dst)
{
    const float* src = dayW + (size_t)(*idx) * 512 * 512;
    for (long i = blockIdx.x * 256L + threadIdx.x; i < 512L * 512; i += gridDim.x * 256L)
        dst[i] = (__bf16)src[i];
}

__global__ __launch_bounds__(256) void k_zero_bf16(__bf16* __restrict__ dst, long n)
{
    for (long i = blockIdx.x * 256L + threadIdx.x; i < n; i += gridDim.x * 256L)
        dst[i] = (__bf16)0.0f;
}

// ---------------------------------------------------------------------------
// Transpose + convert:  src fp32 [R,C] row-major -> dst bf16 [C,R] row-major.
// 32x32 LDS tiles, block (32,8).
// ---------------------------------------------------------------------------
__global__ __launch_bounds__(256) void k_transpose_cvt(const float* __restrict__ src,
                                                       __bf16* __restrict__ dst,
                                                       int R, int C)
{
    __shared__ float tile[32][33];
    const int c0 = blockIdx.x * 32, r0 = blockIdx.y * 32;
    const int tx = threadIdx.x, ty = threadIdx.y;
#pragma unroll
    for (int i = ty; i < 32; i += 8) {
        const int r = r0 + i, c = c0 + tx;
        tile[i][tx] = (r < R && c < C) ? src[(size_t)r * C + c] : 0.0f;
    }
    __syncthreads();
#pragma unroll
    for (int i = ty; i < 32; i += 8) {
        const int c = c0 + i, r = r0 + tx;
        if (c < C && r < R) dst[(size_t)c * R + r] = (__bf16)tile[tx][i];
    }
}

// ---------------------------------------------------------------------------
// Sequential GRU recurrence, one persistent workgroup per direction.
// 512 threads; thread t owns gates (r,z,n) index t. h broadcast via LDS.
// Wh read directly from fp32 input (coalesced across t; 3MB -> L2 resident).
// Writes bf16 [fwd|bwd] feature concat.
// ---------------------------------------------------------------------------
__global__ __launch_bounds__(512) void k_gru(
    const float* __restrict__ gi_f, const float* __restrict__ gi_b,
    const float* __restrict__ Wh_f, const float* __restrict__ bhn_f,
    const float* __restrict__ Wh_b, const float* __restrict__ bhn_b,
    __bf16* __restrict__ out, int S)
{
    const int dir = blockIdx.x;                 // 0 = forward, 1 = backward
    const gcfp gi  = (gcfp)(dir ? gi_b  : gi_f);
    const gcfp Wh  = (gcfp)(dir ? Wh_b  : Wh_f);
    const gcfp bhn = (gcfp)(dir ? bhn_b : bhn_f);
    gbf16p outg = (gbf16p)out;

    __shared__ float hsh[512];
    const int t = threadIdx.x;
    hsh[t] = 0.0f;
    float hreg = 0.0f;
    const float bh = bhn[t];
    __syncthreads();

    for (int step = 0; step < S; ++step) {
        const int s = dir ? (S - 1 - step) : step;
        float ar = 0.0f, az = 0.0f, an = 0.0f;
        unsigned woff = (unsigned)t;
#pragma unroll 8
        for (int d = 0; d < 512; ++d) {
            const float hd = hsh[d];
            ar = fmaf(hd, Wh[woff],        ar);
            az = fmaf(hd, Wh[woff + 512],  az);
            an = fmaf(hd, Wh[woff + 1024], an);
            woff += 1536;
        }
        const unsigned gbase = (unsigned)s * 1536u;
        const float r = 1.0f / (1.0f + __expf(-(gi[gbase + t]        + ar)));
        const float z = 1.0f / (1.0f + __expf(-(gi[gbase + 512 + t]  + az)));
        const float n = tanhf(gi[gbase + 1024 + t] + r * (an + bh));
        const float hn = (1.0f - z) * n + z * hreg;
        __syncthreads();
        hsh[t] = hn;
        hreg   = hn;
        outg[(unsigned)s * 1024u + dir * 512 + t] = (__bf16)hn;
        __syncthreads();
    }
}

// ---------------------------------------------------------------------------
// Decoder: logits = h @ Wdec + bdec ; log_softmax over 41 classes.
// One block (64 threads) per sequence row.
// ---------------------------------------------------------------------------
__global__ __launch_bounds__(64) void k_decoder(const __bf16* __restrict__ h,
                                                const float* __restrict__ Wdec,
                                                const float* __restrict__ bdec,
                                                float* __restrict__ out, int S)
{
    __shared__ float hrow[1024];
    __shared__ float lg[41];
    __shared__ float red[2];
    const int row = blockIdx.x;
    for (int i = threadIdx.x; i < 1024; i += 64)
        hrow[i] = (float)h[(size_t)row * 1024 + i];
    __syncthreads();
    if (threadIdx.x < 41) {
        float a = bdec[threadIdx.x];
#pragma unroll 8
        for (int d = 0; d < 1024; ++d)
            a = fmaf(hrow[d], Wdec[(size_t)d * 41 + threadIdx.x], a);
        lg[threadIdx.x] = a;
    }
    __syncthreads();
    if (threadIdx.x == 0) {
        float mx = lg[0];
        for (int i = 1; i < 41; ++i) mx = fmaxf(mx, lg[i]);
        float se = 0.0f;
        for (int i = 0; i < 41; ++i) se += __expf(lg[i] - mx);
        red[0] = mx; red[1] = __logf(se);
    }
    __syncthreads();
    if (threadIdx.x < 41)
        out[(size_t)row * 41 + threadIdx.x] = lg[threadIdx.x] - red[0] - red[1];
}

// ---------------------------------------------------------------------------
// Host orchestration
// ---------------------------------------------------------------------------
extern "C" void kernel_launch(void* const* d_in, const int* in_sizes, int n_in,
                              void* d_out, int out_size, void* d_ws, size_t ws_size,
                              hipStream_t stream)
{
    (void)in_sizes; (void)n_in; (void)out_size; (void)ws_size;

    const float* x     = (const float*)d_in[0];
    const int*   didx  = (const int*)  d_in[1];
    const float* dayW  = (const float*)d_in[2];
    const float* dayB  = (const float*)d_in[3];
    const float* Wi0f  = (const float*)d_in[4];
    const float* bi0f  = (const float*)d_in[5];
    const float* Wh0f  = (const float*)d_in[6];
    const float* bhn0f = (const float*)d_in[7];
    const float* Wi0b  = (const float*)d_in[8];
    const float* bi0b  = (const float*)d_in[9];
    const float* Wh0b  = (const float*)d_in[10];
    const float* bhn0b = (const float*)d_in[11];
    const float* WiRf  = (const float*)d_in[12];
    const float* biRf  = (const float*)d_in[13];
    const float* WhRf  = (const float*)d_in[14];
    const float* bhnRf = (const float*)d_in[15];
    const float* WiRb  = (const float*)d_in[16];
    const float* biRb  = (const float*)d_in[17];
    const float* WhRb  = (const float*)d_in[18];
    const float* bhnRb = (const float*)d_in[19];
    const float* Wdec  = (const float*)d_in[20];
    const float* bdec  = (const float*)d_in[21];
    float* out = (float*)d_out;

    // ---- workspace carve-up ------------------------------------------------
    char*  ws  = (char*)d_ws;
    size_t off = 0;
    auto alloc = [&](size_t bytes) -> char* {
        char* p = ws + off;
        off = (off + bytes + 255) & ~(size_t)255;
        return p;
    };
    __bf16* xbf  = (__bf16*)alloc(8192ull * 512 * 2);     // x in bf16
    __bf16* wday = (__bf16*)alloc(512ull * 512 * 2);      // day W bf16
    __bf16* hpad = (__bf16*)alloc(8223ull * 512 * 2);     // 31 zero rows + 8192
    __bf16* wt0f = (__bf16*)alloc(1536ull * 16384 * 2);   // Wi0f^T bf16 (N x K)
    __bf16* wt0b = (__bf16*)alloc(1536ull * 16384 * 2);
    __bf16* wtRf[4]; __bf16* wtRb[4];
    for (int l = 0; l < 4; ++l) wtRf[l] = (__bf16*)alloc(1536ull * 1024 * 2);
    for (int l = 0; l < 4; ++l) wtRb[l] = (__bf16*)alloc(1536ull * 1024 * 2);
    float* gif = (float*)alloc(2048ull * 1536 * 4);
    float* gib = (float*)alloc(2048ull * 1536 * 4);
    __bf16* layA = (__bf16*)alloc(2048ull * 1024 * 2);
    __bf16* layB = (__bf16*)alloc(2048ull * 1024 * 2);

    // ---- weight preprocessing (independent of activations) ----------------
    k_cvt_bf16 <<<2048, 256, 0, stream>>>(x, xbf, 8192L * 512);
    k_cvt_day_w<<<256,  256, 0, stream>>>(dayW, didx, wday);
    k_zero_bf16<<<64,   256, 0, stream>>>(hpad, 31L * 512);
    k_transpose_cvt<<<dim3(48, 512), dim3(32, 8), 0, stream>>>(Wi0f, wt0f, 16384, 1536);
    k_transpose_cvt<<<dim3(48, 512), dim3(32, 8), 0, stream>>>(Wi0b, wt0b, 16384, 1536);
    for (int l = 0; l < 4; ++l) {
        k_transpose_cvt<<<dim3(48, 32), dim3(32, 8), 0, stream>>>(
            WiRf + (size_t)l * 1024 * 1536, wtRf[l], 1024, 1536);
        k_transpose_cvt<<<dim3(48, 32), dim3(32, 8), 0, stream>>>(
            WiRb + (size_t)l * 1024 * 1536, wtRb[l], 1024, 1536);
    }

    // ---- day transform: softsign(x @ Wday^T + bday) -> hpad[31:] (bf16) ----
    k_gemm<0, 0><<<dim3(64, 4), 128, 0, stream>>>(
        xbf, wday, dayB, didx, hpad + 31 * 512,
        /*N*/512, /*K*/512, /*lda*/512, /*ldk*/512);

    // ---- layer 0 input gates: im2col-free windowed GEMM, K = 16384 ---------
    k_gemm<1, 1><<<dim3(16, 12), 128, 0, stream>>>(
        hpad, wt0f, bi0f, nullptr, gif, 1536, 16384, 512, 16384);
    k_gemm<1, 1><<<dim3(16, 12), 128, 0, stream>>>(
        hpad, wt0b, bi0b, nullptr, gib, 1536, 16384, 512, 16384);
    k_gru<<<2, 512, 0, stream>>>(gif, gib, Wh0f, bhn0f, Wh0b, bhn0b, layA, 2048);

    // ---- layers 1..4 -------------------------------------------------------
    __bf16* cur = layA;
    __bf16* nxt = layB;
    for (int l = 0; l < 4; ++l) {
        k_gemm<0, 1><<<dim3(16, 12), 128, 0, stream>>>(
            cur, wtRf[l], biRf + (size_t)l * 1536, nullptr, gif,
            1536, 1024, 1024, 1024);
        k_gemm<0, 1><<<dim3(16, 12), 128, 0, stream>>>(
            cur, wtRb[l], biRb + (size_t)l * 1536, nullptr, gib,
            1536, 1024, 1024, 1024);
        k_gru<<<2, 512, 0, stream>>>(
            gif, gib,
            WhRf + (size_t)l * 512 * 1536, bhnRf + (size_t)l * 512,
            WhRb + (size_t)l * 512 * 1536, bhnRb + (size_t)l * 512,
            nxt, 2048);
        __bf16* t = cur; cur = nxt; nxt = t;
    }

    // ---- decoder + log_softmax --------------------------------------------
    k_decoder<<<2048, 64, 0, stream>>>(cur, Wdec, bdec, out, 2048);
}
